// CausalSelfAttention_5145370820674
// MI455X (gfx1250) — matrix operand order repaired and checked by reading, not compiled, not run on hardware
//
#include <hip/hip_runtime.h>
#include <hip/hip_bf16.h>

typedef __bf16 bf16;
typedef __attribute__((ext_vector_type(16))) __bf16 v16bf;
typedef __attribute__((ext_vector_type(8)))  float  v8f;

typedef unsigned int u32;
typedef u32 __attribute__((ext_vector_type(4))) u32x4;
typedef int __attribute__((ext_vector_type(4))) i32x4;
typedef int __attribute__((ext_vector_type(8))) i32x8;

#define D_MODEL 1024
#define N_HEADS 16
#define HEAD_DIM 64
#define SEQ 2048
#define BATCH 4
#define MROWS (BATCH * SEQ) /* 8192 */

#if defined(__HIP_DEVICE_COMPILE__) && __has_builtin(__builtin_amdgcn_tensor_load_to_lds)
#define USE_TDM 1
#else
#define USE_TDM 0
#endif

union FragBf { v16bf v; uint4 q[2]; };
union FragF  { v8f v; float f[8]; };

__device__ __forceinline__ v8f wmma_bf16(const FragBf& a, const FragBf& b, v8f c) {
  return __builtin_amdgcn_wmma_f32_16x16x32_bf16(false, a.v, false, b.v,
                                                 (short)0, c, false, false);
}

#if USE_TDM
// 2-D tile DMA: global (row-major, elem stride0 between rows) -> LDS packed.
// D# per CDNA5 ISA: group0 = {count=1, lds_addr,
// global_addr, type=2}; group1 = {data_size=2B, tensor dims (huge, no OOB),
// tile_d0 x tile_d1, dim0 stride}; groups 2/3 zero (tile_dim3/4 = 0 unused).
__device__ __forceinline__ void tdm_load_2d(u32 lds_off, const bf16* gptr,
                                            int tile_d0, int tile_d1,
                                            int stride0) {
  unsigned long long ga = (unsigned long long)(size_t)gptr;
  const int td0 = 1 << 20, td1 = 1 << 20;
  u32x4 g0;
  g0[0] = 1u;                                        // count=1, user mode
  g0[1] = lds_off;                                   // lds_addr
  g0[2] = (u32)ga;                                   // global_addr[31:0]
  g0[3] = (u32)((ga >> 32) & 0x01FFFFFFu) | (2u << 30);  // addr[56:32]|type=2
  i32x8 g1;
  g1[0] = (1 << 16);                                 // data_size=2B, mask=0
  g1[1] = (td0 & 0xFFFF) << 16;                      // tensor_dim0[15:0]
  g1[2] = ((td0 >> 16) & 0xFFFF) | ((td1 & 0xFFFF) << 16);
  g1[3] = ((td1 >> 16) & 0xFFFF) | ((tile_d0 & 0xFFFF) << 16);
  g1[4] = (tile_d1 & 0xFFFF);                        // tile_dim1, tile_dim2=0
  g1[5] = stride0;                                   // tensor_dim0_stride lo32
  g1[6] = 0;
  g1[7] = 0;
  i32x4 z4 = {0, 0, 0, 0};
#if __clang_major__ >= 23
  i32x8 z8 = {0, 0, 0, 0, 0, 0, 0, 0};
  __builtin_amdgcn_tensor_load_to_lds(g0, g1, z4, z4, z8, 0);
#else
  __builtin_amdgcn_tensor_load_to_lds(g0, g1, z4, z4, 0);
#endif
}
#endif

// ---------------------------------------------------------------- converts --
__global__ void cvt_f32_to_bf16(const float* __restrict__ src,
                                bf16* __restrict__ dst, int n) {
  int i = blockIdx.x * blockDim.x + threadIdx.x;
  int stride = gridDim.x * blockDim.x;
  for (; i < n; i += stride) dst[i] = (bf16)src[i];
}

// ------------------------------------------------------------- QKV GEMM -----
// C[8192,3072] = Xb[8192,1024] @ Wb[1024,3072] + bias; scatter bf16 -> Q/K/V
__launch_bounds__(256)
__global__ void gemm_qkv(const bf16* __restrict__ A, const bf16* __restrict__ W,
                         const float* __restrict__ bias,
                         bf16* __restrict__ Qb, bf16* __restrict__ Kb,
                         bf16* __restrict__ Vb) {
  const int N = 3 * D_MODEL, K = D_MODEL;
  __shared__ bf16 sA[128][64];   // [m][k]
  __shared__ bf16 sBt[128][64];  // [n][k] (transposed on store)
  const int tid = threadIdx.x;
  const int lane = tid & 31, half = (lane >> 4) & 1, l15 = lane & 15;
  const int wid = tid >> 5;
  const int waveM = wid & 3, waveN = wid >> 2;
  const int blockM = blockIdx.x * 128;
  const int blockN = blockIdx.y * 128;

  FragF acc[2][4];
  #pragma unroll
  for (int ms = 0; ms < 2; ms++)
    #pragma unroll
    for (int ns = 0; ns < 4; ns++)
      #pragma unroll
      for (int j = 0; j < 8; j++) acc[ms][ns].f[j] = 0.0f;

  for (int k0 = 0; k0 < K; k0 += 64) {
#if USE_TDM
    if (wid == 0)  // TDM: A tile 128 rows x 64 cols, row pitch K
      tdm_load_2d((u32)(size_t)&sA[0][0], A + (size_t)blockM * K + k0,
                  64, 128, K);
#else
    {  // A tile: 128x64 bf16, 2 threads/row
      int row = tid >> 1, kof = (tid & 1) * 32;
      const uint4* g = (const uint4*)(A + (size_t)(blockM + row) * K + k0 + kof);
      uint4* s = (uint4*)&sA[row][kof];
      s[0] = g[0]; s[1] = g[1]; s[2] = g[2]; s[3] = g[3];
    }
#endif
    {  // W tile: 64x128 bf16, stored transposed [n][k]
      int kr = tid >> 2, nof = (tid & 3) * 32;
      const bf16* g = W + (size_t)(k0 + kr) * N + blockN + nof;
      bf16 tmp[32];
      uint4* tq = (uint4*)tmp;
      const uint4* gq = (const uint4*)g;
      tq[0] = gq[0]; tq[1] = gq[1]; tq[2] = gq[2]; tq[3] = gq[3];
      #pragma unroll
      for (int i = 0; i < 32; i++) sBt[nof + i][kr] = tmp[i];
    }
#if USE_TDM
    if (wid == 0) __builtin_amdgcn_s_wait_tensorcnt(0);
#endif
    __syncthreads();
    #pragma unroll
    for (int kb = 0; kb < 2; kb++) {
      FragBf a[2], b[4];
      #pragma unroll
      for (int ms = 0; ms < 2; ms++) {
        const bf16* p = &sA[waveM * 32 + ms * 16 + l15][kb * 32 + half * 8];
        a[ms].q[0] = *(const uint4*)p;
        a[ms].q[1] = *(const uint4*)(p + 16);
      }
      #pragma unroll
      for (int ns = 0; ns < 4; ns++) {
        const bf16* p = &sBt[waveN * 64 + ns * 16 + l15][kb * 32 + half * 8];
        b[ns].q[0] = *(const uint4*)p;
        b[ns].q[1] = *(const uint4*)(p + 16);
      }
      #pragma unroll
      for (int ms = 0; ms < 2; ms++)
        #pragma unroll
        for (int ns = 0; ns < 4; ns++)
          acc[ms][ns].v = wmma_bf16(a[ms], b[ns], acc[ms][ns].v);
    }
    __syncthreads();
  }

  #pragma unroll
  for (int ns = 0; ns < 4; ns++) {
    int n = blockN + waveN * 64 + ns * 16 + l15;
    float bv = bias[n];
    int sel = n >> 10, n1 = n & 1023;
    int h = n1 >> 6, d = n1 & 63;
    bf16* dst = (sel == 0) ? Qb : ((sel == 1) ? Kb : Vb);
    #pragma unroll
    for (int ms = 0; ms < 2; ms++)
      #pragma unroll
      for (int j = 0; j < 8; j++) {
        int m = blockM + waveM * 32 + ms * 16 + j + half * 8;
        int bz = m >> 11, t = m & (SEQ - 1);
        float v = acc[ms][ns].f[j] + bv;
        dst[((size_t)(bz * N_HEADS + h) * SEQ + t) * HEAD_DIM + d] = (bf16)v;
      }
  }
}

// ------------------------------------------------------- flash attention ---
// 1 block = (b,h) x 64 query rows; 4 waves x 16 rows each.
__launch_bounds__(128)
__global__ void attn(const bf16* __restrict__ Qg, const bf16* __restrict__ Kg,
                     const bf16* __restrict__ Vg, bf16* __restrict__ Og) {
  __shared__ bf16 sK[64][64];       // [key][d]
  __shared__ bf16 sVt[64][64];      // [d][key]
  __shared__ bf16 sP[4][16][64];    // per-wave P scratch [row][key]
  const int tid = threadIdx.x;
  const int lane = tid & 31, half = (lane >> 4) & 1, l15 = lane & 15;
  const int wid = tid >> 5;
  const int gid = blockIdx.x;
  const int qt = gid & 31;          // T/64 = 32 tiles
  const int bh = gid >> 5;          // 0..63
  const int qbase = qt * 64;
  const int qrow0 = qbase + wid * 16;
  const bf16* Qp = Qg + (size_t)bh * SEQ * HEAD_DIM;
  const bf16* Kp = Kg + (size_t)bh * SEQ * HEAD_DIM;
  const bf16* Vp = Vg + (size_t)bh * SEQ * HEAD_DIM;

  FragBf qf[2];
  #pragma unroll
  for (int db = 0; db < 2; db++) {
    const bf16* p = Qp + (size_t)(qrow0 + l15) * HEAD_DIM + db * 32 + half * 8;
    qf[db].q[0] = *(const uint4*)p;
    qf[db].q[1] = *(const uint4*)(p + 16);
  }

  FragF acc[4];
  float mrow[8], lrow[8];
  #pragma unroll
  for (int j = 0; j < 8; j++) { mrow[j] = -1e30f; lrow[j] = 0.0f; }
  #pragma unroll
  for (int ds = 0; ds < 4; ds++)
    #pragma unroll
    for (int j = 0; j < 8; j++) acc[ds].f[j] = 0.0f;

  const float scale = 0.125f;  // 1/sqrt(64)

  for (int kvt = 0; kvt <= qt; kvt++) {
    const int kbase = kvt * 64;
#if USE_TDM
    if (wid == 0)  // TDM: K tile 64x64, row pitch 64 (contiguous)
      tdm_load_2d((u32)(size_t)&sK[0][0], Kp + (size_t)kbase * HEAD_DIM,
                  64, 64, HEAD_DIM);
#else
    {  // K tile -> LDS [key][d]
      int row = tid >> 1, dof = (tid & 1) * 32;
      const uint4* g = (const uint4*)(Kp + (size_t)(kbase + row) * HEAD_DIM + dof);
      uint4* dsts = (uint4*)&sK[row][dof];
      dsts[0] = g[0]; dsts[1] = g[1]; dsts[2] = g[2]; dsts[3] = g[3];
    }
#endif
    {  // V tile -> LDS transposed [d][key]
      int kr = tid >> 1, dof = (tid & 1) * 32;
      const uint4* g = (const uint4*)(Vp + (size_t)(kbase + kr) * HEAD_DIM + dof);
      bf16 tmp[32];
      uint4* tq = (uint4*)tmp;
      tq[0] = g[0]; tq[1] = g[1]; tq[2] = g[2]; tq[3] = g[3];
      #pragma unroll
      for (int i = 0; i < 32; i++) sVt[dof + i][kr] = tmp[i];
    }
    if (kvt < qt) {  // prefetch next K/V tile (L2 resident; hide WGP$ latency)
      const bf16* nk = Kp + (size_t)(kbase + 64 + (tid >> 1)) * HEAD_DIM + (tid & 1) * 32;
      const bf16* nv = Vp + (size_t)(kbase + 64 + (tid >> 1)) * HEAD_DIM + (tid & 1) * 32;
      __builtin_prefetch(nk, 0, 1);
      __builtin_prefetch(nv, 0, 1);
    }
#if USE_TDM
    if (wid == 0) __builtin_amdgcn_s_wait_tensorcnt(0);
#endif
    __syncthreads();

    // S = Q K^T  (4 n-subtiles x 2 d-blocks)
    FragF s[4];
    #pragma unroll
    for (int ns = 0; ns < 4; ns++) {
      v8f c = {};
      #pragma unroll
      for (int db = 0; db < 2; db++) {
        FragBf b;
        const bf16* p = &sK[ns * 16 + l15][db * 32 + half * 8];
        b.q[0] = *(const uint4*)p; b.q[1] = *(const uint4*)(p + 16);
        c = wmma_bf16(qf[db], b, c);
      }
      s[ns].v = c;
    }

    const bool diag = (kvt == qt);
    #pragma unroll
    for (int ns = 0; ns < 4; ns++) {
      int colg = kbase + ns * 16 + l15;
      #pragma unroll
      for (int j = 0; j < 8; j++) {
        float sv = s[ns].f[j] * scale;
        int rowg = qrow0 + j + half * 8;
        if (diag && colg > rowg) sv = -1e30f;
        s[ns].f[j] = sv;
      }
    }

    // online softmax (rows live in 16-lane halves; butterfly stays in-half)
    float alpha[8];
    #pragma unroll
    for (int j = 0; j < 8; j++) {
      float mt = fmaxf(fmaxf(s[0].f[j], s[1].f[j]), fmaxf(s[2].f[j], s[3].f[j]));
      #pragma unroll
      for (int off = 1; off < 16; off <<= 1) mt = fmaxf(mt, __shfl_xor(mt, off, 32));
      float mnew = fmaxf(mrow[j], mt);
      alpha[j] = __expf(mrow[j] - mnew);
      mrow[j] = mnew;
    }
    #pragma unroll
    for (int j = 0; j < 8; j++) {
      float rs = 0.0f;
      #pragma unroll
      for (int ns = 0; ns < 4; ns++) {
        float p = __expf(s[ns].f[j] - mrow[j]);
        s[ns].f[j] = p;
        rs += p;
      }
      #pragma unroll
      for (int off = 1; off < 16; off <<= 1) rs += __shfl_xor(rs, off, 32);
      lrow[j] = lrow[j] * alpha[j] + rs;
    }
    #pragma unroll
    for (int ds = 0; ds < 4; ds++)
      #pragma unroll
      for (int j = 0; j < 8; j++) acc[ds].f[j] *= alpha[j];

    // P -> per-wave LDS scratch (re-layout C-frag -> A-frag), bf16
    #pragma unroll
    for (int ns = 0; ns < 4; ns++)
      #pragma unroll
      for (int j = 0; j < 8; j++)
        sP[wid][j + half * 8][ns * 16 + l15] = (bf16)s[ns].f[j];
    asm volatile("s_wait_dscnt 0" ::: "memory");

    FragBf pa[2];
    #pragma unroll
    for (int kb = 0; kb < 2; kb++) {
      const bf16* p = &sP[wid][l15][kb * 32 + half * 8];
      pa[kb].q[0] = *(const uint4*)p;
      pa[kb].q[1] = *(const uint4*)(p + 16);
    }
    #pragma unroll
    for (int ds = 0; ds < 4; ds++)
      #pragma unroll
      for (int kb = 0; kb < 2; kb++) {
        FragBf b;
        const bf16* p = &sVt[ds * 16 + l15][kb * 32 + half * 8];
        b.q[0] = *(const uint4*)p; b.q[1] = *(const uint4*)(p + 16);
        acc[ds].v = wmma_bf16(pa[kb], b, acc[ds].v);
      }
    __syncthreads();
  }

  // normalize + write merged-head bf16 output [B*T, C]
  const int bb = bh >> 4, h = bh & 15;
  #pragma unroll
  for (int j = 0; j < 8; j++) {
    float inv = 1.0f / lrow[j];
    int rowg = bb * SEQ + qrow0 + j + half * 8;
    #pragma unroll
    for (int ds = 0; ds < 4; ds++) {
      int col = h * HEAD_DIM + ds * 16 + l15;
      Og[(size_t)rowg * D_MODEL + col] = (bf16)(acc[ds].f[j] * inv);
    }
  }
}

// ------------------------------------------------------------- proj GEMM ---
__launch_bounds__(256)
__global__ void gemm_proj(const bf16* __restrict__ A, const bf16* __restrict__ W,
                          const float* __restrict__ bias, float* __restrict__ out) {
  const int N = D_MODEL, K = D_MODEL;
  __shared__ bf16 sA[128][64];
  __shared__ bf16 sBt[128][64];
  const int tid = threadIdx.x;
  const int lane = tid & 31, half = (lane >> 4) & 1, l15 = lane & 15;
  const int wid = tid >> 5;
  const int waveM = wid & 3, waveN = wid >> 2;
  const int blockM = blockIdx.x * 128;
  const int blockN = blockIdx.y * 128;

  FragF acc[2][4];
  #pragma unroll
  for (int ms = 0; ms < 2; ms++)
    #pragma unroll
    for (int ns = 0; ns < 4; ns++)
      #pragma unroll
      for (int j = 0; j < 8; j++) acc[ms][ns].f[j] = 0.0f;

  for (int k0 = 0; k0 < K; k0 += 64) {
#if USE_TDM
    if (wid == 0)
      tdm_load_2d((u32)(size_t)&sA[0][0], A + (size_t)blockM * K + k0,
                  64, 128, K);
#else
    {
      int row = tid >> 1, kof = (tid & 1) * 32;
      const uint4* g = (const uint4*)(A + (size_t)(blockM + row) * K + k0 + kof);
      uint4* s = (uint4*)&sA[row][kof];
      s[0] = g[0]; s[1] = g[1]; s[2] = g[2]; s[3] = g[3];
    }
#endif
    {
      int kr = tid >> 2, nof = (tid & 3) * 32;
      const bf16* g = W + (size_t)(k0 + kr) * N + blockN + nof;
      bf16 tmp[32];
      uint4* tq = (uint4*)tmp;
      const uint4* gq = (const uint4*)g;
      tq[0] = gq[0]; tq[1] = gq[1]; tq[2] = gq[2]; tq[3] = gq[3];
      #pragma unroll
      for (int i = 0; i < 32; i++) sBt[nof + i][kr] = tmp[i];
    }
#if USE_TDM
    if (wid == 0) __builtin_amdgcn_s_wait_tensorcnt(0);
#endif
    __syncthreads();
    #pragma unroll
    for (int kb = 0; kb < 2; kb++) {
      FragBf a[2], b[4];
      #pragma unroll
      for (int ms = 0; ms < 2; ms++) {
        const bf16* p = &sA[waveM * 32 + ms * 16 + l15][kb * 32 + half * 8];
        a[ms].q[0] = *(const uint4*)p;
        a[ms].q[1] = *(const uint4*)(p + 16);
      }
      #pragma unroll
      for (int ns = 0; ns < 4; ns++) {
        const bf16* p = &sBt[waveN * 64 + ns * 16 + l15][kb * 32 + half * 8];
        b[ns].q[0] = *(const uint4*)p;
        b[ns].q[1] = *(const uint4*)(p + 16);
      }
      #pragma unroll
      for (int ms = 0; ms < 2; ms++)
        #pragma unroll
        for (int ns = 0; ns < 4; ns++)
          acc[ms][ns].v = wmma_bf16(a[ms], b[ns], acc[ms][ns].v);
    }
    __syncthreads();
  }

  #pragma unroll
  for (int ns = 0; ns < 4; ns++) {
    int n = blockN + waveN * 64 + ns * 16 + l15;
    float bv = bias[n];
    #pragma unroll
    for (int ms = 0; ms < 2; ms++)
      #pragma unroll
      for (int j = 0; j < 8; j++) {
        int m = blockM + waveM * 32 + ms * 16 + j + half * 8;
        out[(size_t)m * N + n] = acc[ms][ns].f[j] + bv;
      }
  }
}

// --------------------------------------------------------------- launcher --
extern "C" void kernel_launch(void* const* d_in, const int* in_sizes, int n_in,
                              void* d_out, int out_size, void* d_ws, size_t ws_size,
                              hipStream_t stream) {
  (void)in_sizes; (void)n_in; (void)out_size; (void)ws_size;
  const float* x     = (const float*)d_in[0];
  const float* Wqkv  = (const float*)d_in[1];
  const float* bqkv  = (const float*)d_in[2];
  const float* Wproj = (const float*)d_in[3];
  const float* bproj = (const float*)d_in[4];
  float* out = (float*)d_out;

  char* ws = (char*)d_ws;
  size_t off = 0;
  auto alloc = [&](size_t bytes) -> void* {
    void* p = ws + off;
    off += (bytes + 255) & ~(size_t)255;
    return p;
  };
  bf16* xb     = (bf16*)alloc((size_t)MROWS * D_MODEL * 2);
  bf16* wqkvb  = (bf16*)alloc((size_t)D_MODEL * 3 * D_MODEL * 2);
  bf16* wprojb = (bf16*)alloc((size_t)D_MODEL * D_MODEL * 2);
  bf16* Qb     = (bf16*)alloc((size_t)MROWS * D_MODEL * 2);
  bf16* Kb     = (bf16*)alloc((size_t)MROWS * D_MODEL * 2);
  bf16* Vb     = (bf16*)alloc((size_t)MROWS * D_MODEL * 2);
  bf16* Ob     = (bf16*)alloc((size_t)MROWS * D_MODEL * 2);

  cvt_f32_to_bf16<<<2048, 256, 0, stream>>>(x, xb, MROWS * D_MODEL);
  cvt_f32_to_bf16<<<1024, 256, 0, stream>>>(Wqkv, wqkvb, D_MODEL * 3 * D_MODEL);
  cvt_f32_to_bf16<<<512, 256, 0, stream>>>(Wproj, wprojb, D_MODEL * D_MODEL);

  gemm_qkv<<<dim3(MROWS / 128, 3 * D_MODEL / 128), 256, 0, stream>>>(
      xb, wqkvb, bqkv, Qb, Kb, Vb);

  attn<<<BATCH * N_HEADS * (SEQ / 64), 128, 0, stream>>>(Qb, Kb, Vb, Ob);

  gemm_proj<<<dim3(MROWS / 128, D_MODEL / 128), 256, 0, stream>>>(
      Ob, wprojb, bproj, out);
}
